// StackedZOHModel_38336878084732
// MI455X (gfx1250) — matrix operand order repaired and checked by reading, not compiled, not run on hardware
//
#include <hip/hip_runtime.h>
#include <hip/hip_bf16.h>

#define HDIM 512
#define TDIM 104
#define BDIM 512
#define DEPTHN 5
#define NCLS 5
#define TPADX 114   // 2 left-pad + 104 + pad to cover t0+n+kk (max 113)
#define NT 7        // 7 n-tiles of 16 -> 112 columns

typedef __bf16 bf16_t;
typedef __attribute__((ext_vector_type(16))) bf16_t v16bf;
typedef __attribute__((ext_vector_type(16))) unsigned short us16;
typedef __attribute__((ext_vector_type(8)))  unsigned short us8;
typedef __attribute__((ext_vector_type(8)))  float v8f;

static __device__ __forceinline__ unsigned short f2bf(float f) {
  unsigned int u = __float_as_uint(f);
  unsigned int r = u + 0x7FFFu + ((u >> 16) & 1u);   // RNE
  return (unsigned short)(r >> 16);
}

static __device__ __forceinline__ us16 cat16(us8 lo, us8 hi) {
  return __builtin_shufflevector(lo, hi, 0,1,2,3,4,5,6,7,8,9,10,11,12,13,14,15);
}

// CDNA5 async global->LDS copy (ASYNCcnt-tracked, no VGPR data path).
// Inline asm: portable across ROCm-7.2 / amdgpu-toolchain builtin arity skew.
static __device__ __forceinline__ void async_copy_b64(unsigned lds_byte_addr, const void* gaddr) {
  asm volatile("global_load_async_to_lds_b64 %0, %1, off"
               :: "v"(lds_byte_addr), "v"(gaddr) : "memory");
}
static __device__ __forceinline__ void async_wait0() {
  asm volatile("s_wait_asynccnt 0x0" ::: "memory");
}

// ---------------------------------------------------------------------------
// Layer 0: 1->H causal conv (tiny), raw (pre-BN) output
// ---------------------------------------------------------------------------
__global__ void conv0_kernel(const float* __restrict__ x, const float* __restrict__ w0,
                             const float* __restrict__ b0, float* __restrict__ out) {
  __shared__ float xs[TDIM];
  int b = blockIdx.x;
  for (int i = threadIdx.x; i < TDIM; i += blockDim.x) xs[i] = x[b * TDIM + i];
  __syncthreads();
  for (int idx = threadIdx.x; idx < HDIM * TDIM; idx += blockDim.x) {
    int c = idx / TDIM, t = idx % TDIM;
    float acc = b0[c];
#pragma unroll
    for (int k = 0; k < 3; ++k) {
      int tt = t + k - 2;
      if (tt >= 0) acc += w0[c * 3 + k] * xs[tt];
    }
    out[((size_t)b * HDIM + c) * TDIM + t] = acc;
  }
}

// ---------------------------------------------------------------------------
// BatchNorm batch statistics over [B, T] per channel -> fused scale/shift
// ---------------------------------------------------------------------------
__global__ void bn_stats_kernel(const float* __restrict__ act, const float* __restrict__ gamma,
                                const float* __restrict__ beta, float* __restrict__ scale,
                                float* __restrict__ shift) {
  int c = blockIdx.x;
  __shared__ float s1[256], s2[256];
  float a = 0.f, q = 0.f;
  for (int b = threadIdx.x; b < BDIM; b += blockDim.x) {
    const float* p = act + ((size_t)b * HDIM + c) * TDIM;
    for (int t = 0; t < TDIM; t += 4) {
      float4 v = *(const float4*)(p + t);
      a += v.x + v.y + v.z + v.w;
      q += v.x * v.x + v.y * v.y + v.z * v.z + v.w * v.w;
    }
  }
  s1[threadIdx.x] = a; s2[threadIdx.x] = q; __syncthreads();
  for (int s = 128; s > 0; s >>= 1) {
    if (threadIdx.x < s) { s1[threadIdx.x] += s1[threadIdx.x + s]; s2[threadIdx.x] += s2[threadIdx.x + s]; }
    __syncthreads();
  }
  if (threadIdx.x == 0) {
    float n = (float)(BDIM * TDIM);
    float m = s1[0] / n;
    float v = s2[0] / n - m * m;
    float sc = gamma[c] * rsqrtf(v + 1e-5f);
    scale[c] = sc; shift[c] = beta[c] - m * sc;
  }
}

// ---------------------------------------------------------------------------
// Convert conv weights [4][co][ci][3] fp32 -> [4][kk][co][ci] bf16
// ---------------------------------------------------------------------------
__global__ void cvtw_kernel(const float* __restrict__ w, unsigned short* __restrict__ wre, int total) {
  int idx = blockIdx.x * blockDim.x + threadIdx.x;
  if (idx >= total) return;
  int ci = idx & 511;
  int rest = idx >> 9;
  int co = rest & 511;
  int rest2 = rest >> 9;
  int kk = rest2 % 3;
  int l = rest2 / 3;
  float v = w[(((size_t)l * HDIM + co) * HDIM + ci) * 3 + kk];
  wre[idx] = f2bf(v);
}

// ---------------------------------------------------------------------------
// Main causal conv layer as WMMA GEMM: out[b,co,t] = bias[co] +
//   sum_{ci,kk} W[co,ci,kk] * relu(scale[ci]*in[b,ci,t+kk-2]+shift[ci])
// 4 waves / 128 threads; each wave owns 2 m-tiles (32 co rows) so every
// B-fragment feeds two WMMAs. Weights staged via global_load_async_to_lds_b64.
// ---------------------------------------------------------------------------
__global__ __launch_bounds__(128) void conv_wmma_kernel(
    const float* __restrict__ in, const unsigned short* __restrict__ wre,
    const float* __restrict__ bias, const float* __restrict__ scale,
    const float* __restrict__ shift, float* __restrict__ out) {
  __shared__ __align__(16) unsigned short lw[3][128][40];  // only [..][0..31] written (64B rows)
  __shared__ __align__(16) unsigned short lx[TPADX][40];   // lx[t+2][ci]
  int b = blockIdx.y;
  int co0 = blockIdx.x * 128;
  int tid = threadIdx.x;
  int wv = tid >> 5, lane = tid & 31, half = lane >> 4, mrow = lane & 15;

  v8f acc[2][NT];
#pragma unroll
  for (int mi = 0; mi < 2; ++mi)
#pragma unroll
    for (int i = 0; i < NT; ++i)
#pragma unroll
      for (int r = 0; r < 8; ++r) acc[mi][i][r] = 0.f;

  const float* inb = in + (size_t)b * HDIM * TDIM;
  unsigned lwbase = (unsigned)(size_t)(&lw[0][0][0]);
  int sub = lane & 7;      // 8-byte chunk within a 64B row
  int rgrp = lane >> 3;    // row within a 4-row group

  for (int ci0 = 0; ci0 < HDIM; ci0 += 32) {
    // ---- async weight staging: wave copies 96 rows, 4 rows / instruction ----
    for (int i = 0; i < 24; ++i) {
      int row = wv * 96 + i * 4 + rgrp;     // flat row in lw: kk*128 + co
      int kk = row >> 7;
      int co = row & 127;
      const unsigned short* g = wre + ((size_t)kk * HDIM + co0 + co) * HDIM + ci0 + sub * 4;
      unsigned laddr = lwbase + (unsigned)(row * 80 + sub * 8);
      async_copy_b64(laddr, (const void*)g);
    }
    if (ci0 + 32 < HDIM) {  // prefetch next ci-block activation rows
      int ci = tid & 31;
      __builtin_prefetch(inb + (size_t)(ci0 + 32 + ci) * TDIM, 0, 1);
    }
    // ---- stage X tile (BN + ReLU fused, bf16, time-transposed) ----
    for (int idx = tid; idx < 32 * TPADX; idx += 128) {
      int ci = idx / TPADX, tp = idx % TPADX;
      float v = 0.f;
      int t = tp - 2;
      if (t >= 0 && t < TDIM) {
        float xv = inb[(size_t)(ci0 + ci) * TDIM + t];
        v = fmaxf(scale[ci0 + ci] * xv + shift[ci0 + ci], 0.f);
      }
      lx[tp][ci] = f2bf(v);
    }
    async_wait0();
    __syncthreads();

#pragma unroll
    for (int kk = 0; kk < 3; ++kk) {
      v16bf af[2];
#pragma unroll
      for (int mi = 0; mi < 2; ++mi) {
        const unsigned short* ap = &lw[kk][wv * 32 + mi * 16 + mrow][0];
        us8 alo = *reinterpret_cast<const us8*>(ap + 8 * half);
        us8 ahi = *reinterpret_cast<const us8*>(ap + 16 + 8 * half);
        af[mi] = __builtin_bit_cast(v16bf, cat16(alo, ahi));
      }
#pragma unroll
      for (int nt = 0; nt < NT; ++nt) {
        const unsigned short* bp = &lx[nt * 16 + mrow + kk][0];
        us8 blo = *reinterpret_cast<const us8*>(bp + 8 * half);
        us8 bhi = *reinterpret_cast<const us8*>(bp + 16 + 8 * half);
        v16bf bfv = __builtin_bit_cast(v16bf, cat16(blo, bhi));
        acc[0][nt] = __builtin_amdgcn_wmma_f32_16x16x32_bf16(
            false, af[0], false, bfv, (short)0, acc[0][nt], false, false);
        acc[1][nt] = __builtin_amdgcn_wmma_f32_16x16x32_bf16(
            false, af[1], false, bfv, (short)0, acc[1][nt], false, false);
      }
    }
    __syncthreads();
  }

#pragma unroll
  for (int mi = 0; mi < 2; ++mi)
#pragma unroll
    for (int nt = 0; nt < NT; ++nt) {
      int t = nt * 16 + mrow;
      if (t < TDIM) {
#pragma unroll
        for (int r = 0; r < 8; ++r) {
          int co = co0 + wv * 32 + mi * 16 + r + 8 * half;
          out[((size_t)b * HDIM + co) * TDIM + t] = acc[mi][nt][r] + bias[co];
        }
      }
    }
}

// ---------------------------------------------------------------------------
// C[M,N] = A[M,K] @ W[N,K]^T + bias[n]*(*bscale)   (bf16 WMMA, fp32 acc)
// ---------------------------------------------------------------------------
__global__ __launch_bounds__(256) void gemm_bt_wmma_kernel(
    const float* __restrict__ A, const float* __restrict__ W,
    const float* __restrict__ bias, const float* __restrict__ bscale,
    float* __restrict__ C, int Mdim, int Ndim, int Kdim) {
  __shared__ __align__(16) unsigned short la[128][40];
  __shared__ __align__(16) unsigned short lb[128][40];
  int m0 = blockIdx.y * 128, n0 = blockIdx.x * 128;
  int tid = threadIdx.x;
  int wv = tid >> 5, lane = tid & 31, half = lane >> 4, mrow = lane & 15;

  v8f acc[8];
#pragma unroll
  for (int i = 0; i < 8; ++i)
#pragma unroll
    for (int r = 0; r < 8; ++r) acc[i][r] = 0.f;

  for (int k0 = 0; k0 < Kdim; k0 += 32) {
    for (int idx = tid; idx < 128 * 32; idx += 256) {
      int m = idx >> 5, kk = idx & 31;
      la[m][kk] = f2bf(A[(size_t)(m0 + m) * Kdim + k0 + kk]);
      lb[m][kk] = f2bf(W[(size_t)(n0 + m) * Kdim + k0 + kk]);
    }
    __syncthreads();
    const unsigned short* ap = &la[wv * 16 + mrow][0];
    us8 alo = *reinterpret_cast<const us8*>(ap + 8 * half);
    us8 ahi = *reinterpret_cast<const us8*>(ap + 16 + 8 * half);
    v16bf af = __builtin_bit_cast(v16bf, cat16(alo, ahi));
#pragma unroll
    for (int nt = 0; nt < 8; ++nt) {
      const unsigned short* bp = &lb[nt * 16 + mrow][0];
      us8 blo = *reinterpret_cast<const us8*>(bp + 8 * half);
      us8 bhi = *reinterpret_cast<const us8*>(bp + 16 + 8 * half);
      v16bf bfv = __builtin_bit_cast(v16bf, cat16(blo, bhi));
      acc[nt] = __builtin_amdgcn_wmma_f32_16x16x32_bf16(
          false, af, false, bfv, (short)0, acc[nt], false, false);
    }
    __syncthreads();
  }
  float bs = bscale ? *bscale : 1.f;
#pragma unroll
  for (int nt = 0; nt < 8; ++nt) {
    int n = n0 + nt * 16 + mrow;
    float bv = bias ? bias[n] * bs : 0.f;
#pragma unroll
    for (int r = 0; r < 8; ++r) {
      int m = m0 + wv * 16 + r + 8 * half;
      C[(size_t)m * Ndim + n] = acc[nt][r] + bv;
    }
  }
}

// ---------------------------------------------------------------------------
// ZOH: X = 0.1*A ; T1 = I + X/4   (Horner seed for expm Taylor)
// ---------------------------------------------------------------------------
__global__ void zoh_prep_kernel(const float* __restrict__ A_all, float* __restrict__ X,
                                float* __restrict__ T1) {
  size_t idx = (size_t)blockIdx.x * blockDim.x + threadIdx.x;
  size_t total = (size_t)DEPTHN * HDIM * HDIM;
  if (idx >= total) return;
  float a = A_all[idx];
  size_t r = (idx >> 9) & 511, c = idx & 511;
  X[idx] = 0.1f * a;
  T1[idx] = 0.025f * a + ((r == c) ? 1.f : 0.f);
}

// batched fp32 GEMM (512x512x512, z-batched), C = alpha * A@B
__global__ __launch_bounds__(256) void gemm_f32_kernel(const float* __restrict__ A,
                                                       const float* __restrict__ B,
                                                       float* __restrict__ C, float alpha) {
  __shared__ float sa[32][33], sb[32][33];
  size_t zoff = (size_t)blockIdx.z * HDIM * HDIM;
  const float* Az = A + zoff;
  const float* Bz = B + zoff;
  float* Cz = C + zoff;
  int tx = threadIdx.x & 15, ty = threadIdx.x >> 4;
  float c00 = 0, c01 = 0, c10 = 0, c11 = 0;
  for (int k0 = 0; k0 < HDIM; k0 += 32) {
    for (int i = threadIdx.x; i < 32 * 32; i += 256) {
      int r = i >> 5, c = i & 31;
      sa[r][c] = Az[(size_t)(blockIdx.y * 32 + r) * HDIM + k0 + c];
      sb[r][c] = Bz[(size_t)(k0 + r) * HDIM + blockIdx.x * 32 + c];
    }
    __syncthreads();
#pragma unroll 8
    for (int k = 0; k < 32; ++k) {
      float a0 = sa[ty * 2][k], a1 = sa[ty * 2 + 1][k];
      float b0 = sb[k][tx * 2], b1 = sb[k][tx * 2 + 1];
      c00 += a0 * b0; c01 += a0 * b1; c10 += a1 * b0; c11 += a1 * b1;
    }
    __syncthreads();
  }
  int row0 = blockIdx.y * 32 + ty * 2, col0 = blockIdx.x * 32 + tx * 2;
  Cz[(size_t)row0 * HDIM + col0] = alpha * c00;
  Cz[(size_t)row0 * HDIM + col0 + 1] = alpha * c01;
  Cz[(size_t)(row0 + 1) * HDIM + col0] = alpha * c10;
  Cz[(size_t)(row0 + 1) * HDIM + col0 + 1] = alpha * c11;
}

__global__ void addI_kernel(float* __restrict__ C) {
  size_t z = blockIdx.x; int i = threadIdx.x;
  C[z * HDIM * HDIM + (size_t)i * HDIM + i] += 1.f;
}

// r = (Abar - I) @ (DELTA * Bv)
__global__ void rhs_kernel(const float* __restrict__ Abar, const float* __restrict__ B_all,
                           float* __restrict__ rhs) {
  int z = blockIdx.x; int i = threadIdx.x;
  __shared__ float db[HDIM];
  db[i] = 0.1f * B_all[z * HDIM + i];
  __syncthreads();
  const float* Ar = Abar + (size_t)z * HDIM * HDIM + (size_t)i * HDIM;
  float s = 0.f;
  for (int j = 0; j < HDIM; ++j) s += Ar[j] * db[j];
  rhs[z * HDIM + i] = s - db[i];
}

// augmented [A + 1e-3 I | r]
__global__ void buildG_kernel(const float* __restrict__ A_all, const float* __restrict__ rhs,
                              float* __restrict__ G) {
  int r = blockIdx.x, z = blockIdx.y;
  float* Gr = G + (size_t)z * HDIM * (HDIM + 1) + (size_t)r * (HDIM + 1);
  const float* Ar = A_all + (size_t)z * HDIM * HDIM + (size_t)r * HDIM;
  for (int c = threadIdx.x; c < HDIM; c += blockDim.x)
    Gr[c] = Ar[c] + ((c == r) ? 1e-3f : 0.f);
  if (threadIdx.x == 0) Gr[HDIM] = rhs[z * HDIM + r];
}

// Gauss-Jordan with partial pivoting; solution ends up in column HDIM
__global__ __launch_bounds__(1024) void gauss_kernel(float* __restrict__ Gall) {
  const int N = HDIM, W = HDIM + 1;
  float* G = Gall + (size_t)blockIdx.x * N * W;
  __shared__ float rv[1024]; __shared__ int ri[1024];
  __shared__ float ff[HDIM]; __shared__ float prow[HDIM + 1];
  __shared__ int pivrow_s; __shared__ float pinv_s;
  int tid = threadIdx.x;
  for (int p = 0; p < N; ++p) {
    float best = -1.f; int bi = p;
    for (int r = p + tid; r < N; r += 1024) {
      float v = fabsf(G[(size_t)r * W + p]);
      if (v > best) { best = v; bi = r; }
    }
    rv[tid] = best; ri[tid] = bi; __syncthreads();
    for (int s = 512; s > 0; s >>= 1) {
      if (tid < s && rv[tid + s] > rv[tid]) { rv[tid] = rv[tid + s]; ri[tid] = ri[tid + s]; }
      __syncthreads();
    }
    if (tid == 0) {
      pivrow_s = ri[0];
      pinv_s = 1.f / G[(size_t)ri[0] * W + p];
    }
    __syncthreads();
    int pr = pivrow_s;
    float pinv = pinv_s;
    for (int c = p + tid; c <= N; c += 1024) {
      float vp = G[(size_t)p * W + c];
      float vr = G[(size_t)pr * W + c];
      float np = ((pr != p) ? vr : vp) * pinv;
      G[(size_t)p * W + c] = np;
      if (pr != p) G[(size_t)pr * W + c] = vp;
      prow[c] = np;
    }
    __syncthreads();
    for (int r = tid; r < N; r += 1024) ff[r] = G[(size_t)r * W + p];
    __syncthreads();
    int len = N - p + 1;
    long total = (long)N * len;
    for (long i2 = tid; i2 < total; i2 += 1024) {
      int r = (int)(i2 / len);
      int c = p + (int)(i2 % len);
      if (r != p) G[(size_t)r * W + c] -= ff[r] * prow[c];
    }
    __syncthreads();
  }
}

// sequential scan: terms_k = C @ Abar^k @ Bbar ; kb = flip(terms)/||terms||
__global__ __launch_bounds__(512) void scan_kernel(const float* __restrict__ Abar,
                                                   const float* __restrict__ Gall,
                                                   const float* __restrict__ C_all,
                                                   float* __restrict__ kb) {
  int z = blockIdx.x, i = threadIdx.x;
  __shared__ float v[HDIM], cv[HDIM], red[HDIM], terms[TDIM];
  const float* Az = Abar + (size_t)z * HDIM * HDIM;
  v[i] = Gall[(size_t)z * HDIM * (HDIM + 1) + (size_t)i * (HDIM + 1) + HDIM];
  cv[i] = C_all[z * HDIM + i];
  __syncthreads();
  for (int k = 0; k < TDIM; ++k) {
    red[i] = cv[i] * v[i]; __syncthreads();
    for (int s = 256; s > 0; s >>= 1) { if (i < s) red[i] += red[i + s]; __syncthreads(); }
    if (i == 0) terms[k] = red[0];
    const float* Ar = Az + (size_t)i * HDIM;
    float nv = 0.f;
    for (int j = 0; j < HDIM; ++j) nv += Ar[j] * v[j];
    __syncthreads();
    v[i] = nv; __syncthreads();
  }
  red[i] = 0.f;
  if (i < TDIM) red[i] = terms[i] * terms[i];
  __syncthreads();
  for (int s = 256; s > 0; s >>= 1) { if (i < s) red[i] += red[i + s]; __syncthreads(); }
  float rn = 1.f / (sqrtf(red[0]) + 1e-6f);
  if (i < TDIM) kb[z * TDIM + i] = terms[TDIM - 1 - i] * rn;
}

__global__ void kprod_kernel(const float* __restrict__ kb, float* __restrict__ kprod) {
  int t = threadIdx.x;
  if (t >= TDIM) return;
  float p = 1.f;
  for (int l = 0; l < DEPTHN; ++l) p *= kb[l * TDIM + t];
  kprod[t] = p;
}

// peK[h] = mean_t kprod[t]*pe(t,h); ksum = mean_t kprod[t]
__global__ void pe_kernel(const float* __restrict__ kprod, float* __restrict__ peK,
                          float* __restrict__ ksum) {
  int h = threadIdx.x;
  __shared__ float kp[TDIM];
  if (h < TDIM) kp[h] = kprod[h];
  __syncthreads();
  int base = h & ~1;
  float div = expf(-logf(10000.f) * (float)base / (float)HDIM);
  float s = 0.f;
  for (int t = 0; t < TDIM; ++t) {
    float ang = (float)t * div;
    float pe = (h & 1) ? cosf(ang) : sinf(ang);
    s += kp[t] * pe;
  }
  peK[h] = s / (float)TDIM;
  if (h == 0) {
    float ks = 0.f;
    for (int t = 0; t < TDIM; ++t) ks += kp[t];
    ksum[0] = ks / (float)TDIM;
  }
}

// g[b,h] = mean_t kprod[t]*relu(scale*act+shift) + peK[h]
__global__ void time_reduce_kernel(const float* __restrict__ act, const float* __restrict__ scale,
                                   const float* __restrict__ shift, const float* __restrict__ kprod,
                                   const float* __restrict__ peK, float* __restrict__ g) {
  int b = blockIdx.x;
  __shared__ float kp[TDIM];
  for (int t = threadIdx.x; t < TDIM; t += blockDim.x) kp[t] = kprod[t];
  __syncthreads();
  for (int h = threadIdx.x; h < HDIM; h += blockDim.x) {
    const float* p = act + ((size_t)b * HDIM + h) * TDIM;
    float sc = scale[h], sh = shift[h], s = 0.f;
    for (int t = 0; t < TDIM; ++t) s += kp[t] * fmaxf(sc * p[t] + sh, 0.f);
    g[(size_t)b * HDIM + h] = s / (float)TDIM + peK[h];
  }
}

// fc BN (batch stats over B per feature) + exact GELU, in place
__global__ void bn_gelu_kernel(float* __restrict__ z, const float* __restrict__ gamma,
                               const float* __restrict__ beta) {
  int o = blockIdx.x;
  __shared__ float s1[256], s2[256];
  float a = 0.f, q = 0.f;
  for (int b = threadIdx.x; b < BDIM; b += 256) {
    float v = z[(size_t)b * HDIM + o]; a += v; q += v * v;
  }
  s1[threadIdx.x] = a; s2[threadIdx.x] = q; __syncthreads();
  for (int s = 128; s > 0; s >>= 1) {
    if (threadIdx.x < s) { s1[threadIdx.x] += s1[threadIdx.x + s]; s2[threadIdx.x] += s2[threadIdx.x + s]; }
    __syncthreads();
  }
  float m = s1[0] / (float)BDIM, var = s2[0] / (float)BDIM - m * m;
  float sc = gamma[o] * rsqrtf(var + 1e-5f), sh = beta[o] - m * sc;
  for (int b = threadIdx.x; b < BDIM; b += 256) {
    size_t idx = (size_t)b * HDIM + o;
    float x = sc * z[idx] + sh;
    z[idx] = 0.5f * x * (1.f + erff(x * 0.70710678f));
  }
}

__global__ void fc2_kernel(const float* __restrict__ z, const float* __restrict__ w2,
                           const float* __restrict__ b2, float* __restrict__ out) {
  int b = blockIdx.x;
  __shared__ float red[256];
  const float* zr = z + (size_t)b * HDIM;
  for (int n = 0; n < NCLS; ++n) {
    float s = 0.f;
    for (int h = threadIdx.x; h < HDIM; h += 256) s += zr[h] * w2[n * HDIM + h];
    red[threadIdx.x] = s; __syncthreads();
    for (int st = 128; st > 0; st >>= 1) {
      if (threadIdx.x < st) red[threadIdx.x] += red[threadIdx.x + st];
      __syncthreads();
    }
    if (threadIdx.x == 0) out[b * NCLS + n] = red[0] + b2[n];
    __syncthreads();
  }
}

// ---------------------------------------------------------------------------
static inline size_t alup(size_t x) { return (x + 255) & ~(size_t)255; }

extern "C" void kernel_launch(void* const* d_in, const int* in_sizes, int n_in,
                              void* d_out, int out_size, void* d_ws, size_t ws_size,
                              hipStream_t stream) {
  const float* x        = (const float*)d_in[0];
  const float* conv_w0  = (const float*)d_in[1];
  const float* conv_b0  = (const float*)d_in[2];
  const float* conv_w   = (const float*)d_in[3];
  const float* conv_b   = (const float*)d_in[4];
  const float* bn_gamma = (const float*)d_in[5];
  const float* bn_beta  = (const float*)d_in[6];
  const float* proj_w   = (const float*)d_in[7];
  const float* proj_b   = (const float*)d_in[8];
  const float* A_all    = (const float*)d_in[9];
  const float* B_all    = (const float*)d_in[10];
  const float* C_all    = (const float*)d_in[11];
  const float* fc1_w    = (const float*)d_in[12];
  const float* fc1_b    = (const float*)d_in[13];
  const float* fcg      = (const float*)d_in[14];
  const float* fcb      = (const float*)d_in[15];
  const float* fc2_w    = (const float*)d_in[16];
  const float* fc2_b    = (const float*)d_in[17];
  float* out = (float*)d_out;

  char* ws = (char*)d_ws;
  size_t off = 0;
  const size_t ACT = (size_t)BDIM * HDIM * TDIM * sizeof(float);
  const size_t MAT = (size_t)DEPTHN * HDIM * HDIM * sizeof(float);
  float* actA = (float*)(ws + off);            off = alup(off + ACT);
  float* actB = (float*)(ws + off);            off = alup(off + ACT);
  unsigned short* wre = (unsigned short*)(ws + off);
  off = alup(off + (size_t)4 * 3 * HDIM * HDIM * sizeof(unsigned short));
  float* scaleB = (float*)(ws + off);          off = alup(off + HDIM * sizeof(float));
  float* shiftB = (float*)(ws + off);          off = alup(off + HDIM * sizeof(float));
  float* Xm   = (float*)(ws + off);            off = alup(off + MAT);
  float* T1m  = (float*)(ws + off);            off = alup(off + MAT);
  float* T2m  = (float*)(ws + off);            off = alup(off + MAT);
  float* Abar = (float*)(ws + off);            off = alup(off + MAT);
  float* rhsB = (float*)(ws + off);            off = alup(off + DEPTHN * HDIM * sizeof(float));
  float* Gm   = (float*)(ws + off);            off = alup(off + (size_t)DEPTHN * HDIM * (HDIM + 1) * sizeof(float));
  float* kbB  = (float*)(ws + off);            off = alup(off + DEPTHN * TDIM * sizeof(float));
  float* kprodB = (float*)(ws + off);          off = alup(off + TDIM * sizeof(float));
  float* peKB = (float*)(ws + off);            off = alup(off + HDIM * sizeof(float));
  float* ksumB = (float*)(ws + off);           off = alup(off + 256);
  float* gB   = (float*)(ws + off);            off = alup(off + (size_t)BDIM * HDIM * sizeof(float));
  float* pooled = (float*)(ws + off);          off = alup(off + (size_t)BDIM * HDIM * sizeof(float));
  float* zB   = (float*)(ws + off);            off = alup(off + (size_t)BDIM * HDIM * sizeof(float));
  (void)ws_size; (void)in_sizes; (void)n_in; (void)out_size;

  // weight precision conversion (bf16, tap-major layout)
  int wtot = 4 * 3 * HDIM * HDIM;
  cvtw_kernel<<<(wtot + 255) / 256, 256, 0, stream>>>(conv_w, wre, wtot);

  // causal CNN stack
  conv0_kernel<<<BDIM, 256, 0, stream>>>(x, conv_w0, conv_b0, actA);
  bn_stats_kernel<<<HDIM, 256, 0, stream>>>(actA, bn_gamma, bn_beta, scaleB, shiftB);
  for (int l = 0; l < 4; ++l) {
    const float* inp = (l & 1) ? actB : actA;
    float* outp = (l & 1) ? actA : actB;
    const unsigned short* wl = wre + (size_t)l * 3 * HDIM * HDIM;
    conv_wmma_kernel<<<dim3(4, BDIM), 128, 0, stream>>>(inp, wl, conv_b + (size_t)l * HDIM,
                                                        scaleB, shiftB, outp);
    bn_stats_kernel<<<HDIM, 256, 0, stream>>>(outp, bn_gamma + (size_t)(l + 1) * HDIM,
                                              bn_beta + (size_t)(l + 1) * HDIM, scaleB, shiftB);
  }
  // final activation is in actA (conv0->A, l0 A->B, l1 B->A, l2 A->B, l3 B->A)

  // ZOH kernels: expm via 4-term Taylor (||0.1A|| ~ 5e-3), vector linear solve
  size_t mt = (size_t)DEPTHN * HDIM * HDIM;
  zoh_prep_kernel<<<(unsigned)((mt + 255) / 256), 256, 0, stream>>>(A_all, Xm, T1m);
  gemm_f32_kernel<<<dim3(16, 16, DEPTHN), 256, 0, stream>>>(Xm, T1m, T2m, 1.f / 3.f);
  addI_kernel<<<DEPTHN, HDIM, 0, stream>>>(T2m);
  gemm_f32_kernel<<<dim3(16, 16, DEPTHN), 256, 0, stream>>>(Xm, T2m, T1m, 0.5f);
  addI_kernel<<<DEPTHN, HDIM, 0, stream>>>(T1m);
  gemm_f32_kernel<<<dim3(16, 16, DEPTHN), 256, 0, stream>>>(Xm, T1m, Abar, 1.f);
  addI_kernel<<<DEPTHN, HDIM, 0, stream>>>(Abar);
  rhs_kernel<<<DEPTHN, HDIM, 0, stream>>>(Abar, B_all, rhsB);
  buildG_kernel<<<dim3(HDIM, DEPTHN), 256, 0, stream>>>(A_all, rhsB, Gm);
  gauss_kernel<<<DEPTHN, 1024, 0, stream>>>(Gm);
  scan_kernel<<<DEPTHN, HDIM, 0, stream>>>(Abar, Gm, C_all, kbB);
  kprod_kernel<<<1, 128, 0, stream>>>(kbB, kprodB);
  pe_kernel<<<1, HDIM, 0, stream>>>(kprodB, peKB, ksumB);

  // kprod-weighted time mean (proj commuted past the mean), then head GEMMs
  time_reduce_kernel<<<BDIM, 256, 0, stream>>>(actA, scaleB, shiftB, kprodB, peKB, gB);
  gemm_bt_wmma_kernel<<<dim3(4, 4), 256, 0, stream>>>(gB, proj_w, proj_b, ksumB, pooled,
                                                      BDIM, HDIM, HDIM);
  gemm_bt_wmma_kernel<<<dim3(4, 4), 256, 0, stream>>>(pooled, fc1_w, fc1_b, nullptr, zB,
                                                      BDIM, HDIM, HDIM);
  bn_gelu_kernel<<<HDIM, 256, 0, stream>>>(zB, fcg, fcb);
  fc2_kernel<<<BDIM, 256, 0, stream>>>(zB, fc2_w, fc2_b, out);
}
